// SegCaps_46832323395612
// MI455X (gfx1250) — matrix operand order repaired and checked by reading, not compiled, not run on hardware
//
#include <hip/hip_runtime.h>
#include <math.h>

typedef _Float16 v4h  __attribute__((ext_vector_type(4)));
typedef _Float16 v8h  __attribute__((ext_vector_type(8)));
typedef _Float16 v16h __attribute__((ext_vector_type(16)));
typedef float    v8f  __attribute__((ext_vector_type(8)));
typedef unsigned int u32x4 __attribute__((ext_vector_type(4)));
typedef int          s32x8 __attribute__((ext_vector_type(8)));
typedef int          s32x4 __attribute__((ext_vector_type(4)));

#define MAX_KPAD 1600   // c3b: Cin=64 -> K=1600 (already multiple of 32)

// ---------------------------------------------------------------------------
// TDM: 1-D contiguous copy global -> LDS (D# groups built per cdna5 ISA 8.3/8.4)
//   group0: count=1 | lds_addr | global_addr | type=2
//   group1: data_size=2B, tensor_dim0 = tile_dim0 = nHalves, no pad/iterate
// 6-arg builtin (clang-23 signature): (g0, g1, g2, g3, g4, cpol)
// ---------------------------------------------------------------------------
__device__ __forceinline__ void tdm_load_1d(const _Float16* gptr, _Float16* lptr,
                                            int nHalves) {
#if __has_builtin(__builtin_amdgcn_tensor_load_to_lds)
    unsigned lds = (unsigned)(size_t)(void*)lptr;
    unsigned long long ga = (unsigned long long)(size_t)gptr;
    u32x4 g0;
    g0[0] = 1u;                                            // count=1, user mode
    g0[1] = lds;                                           // lds_addr
    g0[2] = (unsigned)ga;                                  // global_addr[31:0]
    g0[3] = (unsigned)((ga >> 32) & 0x01FFFFFFu) | (2u << 30);  // addr[56:32] | type=2
    s32x8 g1 = {};
    g1[0] = 0x10000;                                       // data_size = 1 -> 2 bytes
    g1[1] = (int)(((unsigned)nHalves & 0xFFFFu) << 16);    // tensor_dim0[15:0]
    g1[2] = (int)(((unsigned)nHalves >> 16) & 0xFFFFu);    // tensor_dim0[31:16]
    g1[3] = (int)(((unsigned)nHalves & 0xFFFFu) << 16);    // tile_dim0
    s32x4 z4 = {};
    s32x8 z8 = {};
    __builtin_amdgcn_tensor_load_to_lds(g0, g1, z4, z4, z8, 0);
    __builtin_amdgcn_s_wait_tensorcnt(0);
#else
    // fallback: scalar-wave cooperative copy (lane-strided b128)
    int lane = threadIdx.x & 31;
    for (int i = lane * 8; i < nHalves; i += 32 * 8)
        *(v8h*)(lptr + i) = *(const v8h*)(gptr + i);
#endif
}

// ---------------------------------------------------------------------------
// Zero-fill (logits init; graph-capture safe)
// ---------------------------------------------------------------------------
__global__ void fill0(float* __restrict__ p, size_t n) {
    size_t i = (size_t)blockIdx.x * blockDim.x + threadIdx.x;
    if (i < n) p[i] = 0.0f;
}

// fp32 -> f16 elementwise (stem input conversion)
__global__ void cvt_f32_to_f16(const float* __restrict__ x, _Float16* __restrict__ y, size_t n) {
    size_t i = (size_t)blockIdx.x * blockDim.x + threadIdx.x;
    if (i < n) y[i] = (_Float16)x[i];
}

// ---------------------------------------------------------------------------
// fp32 OIHW weights -> f16 [Cout, Kpad], k = (ky*5+kx)*Cin + ci
// ---------------------------------------------------------------------------
__global__ void cvt_weights(const float* __restrict__ w, _Float16* __restrict__ w16,
                            int Cout, int cinShift, int Kpad) {
    int idx = blockIdx.x * blockDim.x + threadIdx.x;
    int total = Cout * Kpad;
    if (idx >= total) return;
    int co = idx / Kpad;
    int k  = idx - co * Kpad;
    int Cin = 1 << cinShift;
    int K   = Cin * 25;
    float v = 0.0f;
    if (k < K) {
        int tap = k >> cinShift;          // ky*5+kx
        int ci  = k & (Cin - 1);
        v = w[((size_t)co * Cin + ci) * 25 + tap];
    }
    w16[idx] = (_Float16)v;
}

// ---------------------------------------------------------------------------
// Implicit-GEMM 5x5 conv / transposed-conv via v_wmma_f32_16x16x32_f16.
//   Block (128 thr = 4 waves): 16 M x 256 N; wave: 16 M x 64 N (4 acc tiles).
//   Weights [16 x Kpad] staged once in LDS via TDM; K-loop is barrier-free:
//     2 x ds_load_b128 (A) + <=8 x global_load_b128 (B) + 4 x v_wmma.
//   Cin >= 16 only: tap index is constant per lane per 32-K chunk.
//   Activations f16 channel-innermost: [b][t][p][ci].
// ---------------------------------------------------------------------------
template <int DECONV>
__global__ __launch_bounds__(128) void conv5x5_wmma(
    const _Float16* __restrict__ in_base, long in_sB, int t0,
    int cinShift, int H, int W,
    const _Float16* __restrict__ w16, int Kpad,
    const float* __restrict__ bias,
    float* __restrict__ outF, _Float16* __restrict__ outH, long out_sB,
    int Cout, int Ho, int Wo, int stride)
{
    __shared__ __align__(16) _Float16 As[16 * MAX_KPAD];

    const int Cin = 1 << cinShift;
    const int K   = Cin * 25;
    const int P   = Ho * Wo;

    const int tid  = threadIdx.x;
    const int lane = tid & 31;
    const int wv   = tid >> 5;
    const int lr   = lane & 15;
    const int hi   = lane >> 4;

    const int mBase = blockIdx.x * 16;
    const int bt = blockIdx.z;
    const int b  = bt / t0;
    const int t  = bt - b * t0;

    const _Float16* inp = in_base + (size_t)b * in_sB + (size_t)t * Cin * H * W;

    // ---- stage the whole weight slab [16 rows x Kpad] once (rows contiguous)
    if (wv == 0)
        tdm_load_1d(w16 + (size_t)mBase * Kpad, As, 16 * Kpad);
    __syncthreads();

    // ---- 4 N-subtiles per wave
    const int nBase = (blockIdx.y * 16 + wv * 4) * 16;
    int ho[4], wo[4];
    bool vn[4];
#pragma unroll
    for (int s = 0; s < 4; ++s) {
        int n = nBase + s * 16 + lr;
        vn[s] = (n < P);
        int h = vn[s] ? n / Wo : 0;
        ho[s] = h;
        wo[s] = vn[s] ? (n - h * Wo) : 0;
    }

    v8f acc[4] = {};
    const _Float16* aRow = As + (size_t)lr * Kpad;

    for (int k0 = 0; k0 < K; k0 += 32) {
        // A fragment (two 16B ds loads), shared by all 4 subtiles
        v16h a;
        {
            v8h a0 = *(const v8h*)(aRow + k0 + hi * 16);
            v8h a1 = *(const v8h*)(aRow + k0 + hi * 16 + 8);
#pragma unroll
            for (int j = 0; j < 8; ++j) { a[j] = a0[j]; a[8 + j] = a1[j]; }
        }

        // per-chunk tap math (constant per lane)
        const int e0  = k0 + hi * 16;
        const int tap = e0 >> cinShift;
        const int cib = e0 & (Cin - 1);
        const bool kok = (tap < 25);
        const int ky = tap / 5;
        const int kx = tap - ky * 5;

#pragma unroll
        for (int s = 0; s < 4; ++s) {
            bool ok = kok && vn[s];
            int hy, wx;
            if (DECONV) {
                int hd = ho[s] + ky - 2, wd = wo[s] + kx - 2;
                ok = ok && (hd >= 0) && (wd >= 0) && !(hd & 1) && !(wd & 1);
                hy = hd >> 1; wx = wd >> 1;
                ok = ok && (hy < H) && (wx < W);
            } else {
                hy = ho[s] * stride + ky - 2;
                wx = wo[s] * stride + kx - 2;
                ok = ok && (hy >= 0) && (hy < H) && (wx >= 0) && (wx < W);
            }
            v16h bf;
            if (ok) {
                const _Float16* bp = inp + ((size_t)(hy * W + wx)) * Cin + cib;
                v8h b0 = *(const v8h*)bp;
                v8h b1 = *(const v8h*)(bp + 8);
#pragma unroll
                for (int j = 0; j < 8; ++j) { bf[j] = b0[j]; bf[8 + j] = b1[j]; }
            } else {
#pragma unroll
                for (int j = 0; j < 16; ++j) bf[j] = (_Float16)0.0f;
            }
            // EXEC is full here (guards above only predicate loads).
            acc[s] = __builtin_amdgcn_wmma_f32_16x16x32_f16(
                         false, a, false, bf, (short)0, acc[s], false, false);
        }
    }

    // ---- store: C/D layout: VGPR r -> M = hi*8 + r ; N = lane&15
#pragma unroll
    for (int s = 0; s < 4; ++s) {
        if (!vn[s]) continue;
        int n = nBase + s * 16 + lr;
        if (outF) {
#pragma unroll
            for (int r = 0; r < 8; ++r) {
                int co = mBase + hi * 8 + r;
                if (co < Cout) {
                    float v = acc[s][r] + (bias ? bias[co] : 0.0f);
                    outF[(size_t)b * out_sB + ((size_t)t * Cout + co) * P + n] = v;
                }
            }
        } else {
#pragma unroll
            for (int r = 0; r < 8; ++r) {
                int co = mBase + hi * 8 + r;
                if (co < Cout) {
                    float v = acc[s][r] + (bias ? bias[co] : 0.0f);
                    outH[(size_t)b * out_sB + ((size_t)t * P + n) * Cout + co] = (_Float16)v;
                }
            }
        }
    }
}

// ---------------------------------------------------------------------------
// Stem conv (Cin=1, Cout=16, stride 1): K=25 -> a SINGLE 16x16x32 WMMA.
// One wave per 16-pixel tile; A read directly from w16 [16 x 32].
// Output f16 [b][p][z=16] (written into cat384 @ t=2).
// ---------------------------------------------------------------------------
__global__ __launch_bounds__(128) void conv_stem_wmma(
    const _Float16* __restrict__ x,      // [b][p], b-stride = P
    const _Float16* __restrict__ w16,    // [16][32] (Kpad=32, zero-padded)
    _Float16* __restrict__ outH, long out_sB,
    int H, int W)
{
    const int P = H * W;
    const int tid  = threadIdx.x;
    const int lane = tid & 31;
    const int wv   = tid >> 5;
    const int lr   = lane & 15;
    const int hi   = lane >> 4;

    const int b = blockIdx.z;
    const _Float16* inp = x + (size_t)b * P;

    const int n  = (blockIdx.y * 4 + wv) * 16 + lr;
    const bool vn = (n < P);
    const int ho = vn ? n / W : 0;
    const int wo = vn ? (n - ho * W) : 0;

    v16h a;
    {
        const _Float16* aRow = w16 + (size_t)lr * 32;
        v8h a0 = *(const v8h*)(aRow + hi * 16);
        v8h a1 = *(const v8h*)(aRow + hi * 16 + 8);
#pragma unroll
        for (int j = 0; j < 8; ++j) { a[j] = a0[j]; a[8 + j] = a1[j]; }
    }

    v16h bf;
#pragma unroll
    for (int j = 0; j < 16; ++j) {
        _Float16 val = (_Float16)0.0f;
        int k = hi * 16 + j;             // tap index (Cin=1)
        if (vn && k < 25) {
            int ky = k / 5;
            int kx = k - ky * 5;
            int hy = ho + ky - 2;
            int wx = wo + kx - 2;
            if (hy >= 0 && hy < H && wx >= 0 && wx < W)
                val = inp[(size_t)hy * W + wx];
        }
        bf[j] = val;
    }

    v8f acc = {};
    acc = __builtin_amdgcn_wmma_f32_16x16x32_f16(
              false, a, false, bf, (short)0, acc, false, false);

    if (!vn) return;
#pragma unroll
    for (int r = 0; r < 8; ++r) {
        int co = hi * 8 + r;
        outH[(size_t)b * out_sB + (size_t)n * 16 + co] = (_Float16)acc[r];
    }
}

// ---------------------------------------------------------------------------
// Routing step 1: c = softmax_u(logits); s = sum_t c*u; v = squash_z(s).
// u: fp32 [B,t0,t1,z1,P]; logits: fp32 [B,t0,t1,P];
// v written f16, z-contiguous view: v_base + b*v_sB + (uu*P + p)*z1 + z
// ---------------------------------------------------------------------------
__global__ void route_reduce(const float* __restrict__ u,
                             const float* __restrict__ logits,
                             _Float16* __restrict__ v_base, long v_sB,
                             int B, int t0, int t1, int z1, int P)
{
    size_t idx = (size_t)blockIdx.x * blockDim.x + threadIdx.x;
    size_t total = (size_t)B * t1 * P;
    if (idx >= total) return;
    int p   = (int)(idx % P);
    int rem = (int)(idx / P);
    int uu  = rem % t1;
    int b   = rem / t1;

    float ct[16];
#pragma unroll
    for (int t = 0; t < 16; ++t) {
        float c = 0.0f;
        if (t < t0) {
            const float* lrow = logits + (((size_t)b * t0 + t) * t1) * P + p;
            float lmax = -1e30f;
#pragma unroll
            for (int j = 0; j < 8; ++j)
                if (j < t1) lmax = fmaxf(lmax, lrow[(size_t)j * P]);
            float sum = 0.0f, mine = 0.0f;
#pragma unroll
            for (int j = 0; j < 8; ++j)
                if (j < t1) {
                    float e = __expf(lrow[(size_t)j * P] - lmax);
                    sum += e;
                    if (j == uu) mine = e;
                }
            c = mine / sum;
        }
        ct[t] = c;
    }

    float sq = 0.0f;
    for (int z = 0; z < z1; ++z) {
        float s = 0.0f;
#pragma unroll
        for (int t = 0; t < 16; ++t)
            if (t < t0)
                s += ct[t] * u[((((size_t)b * t0 + t) * t1 + uu) * z1 + z) * P + p];
        sq += s * s;
    }
    float factor = (sq / (1.0f + sq)) * rsqrtf(sq + 1e-7f);

    for (int z = 0; z < z1; ++z) {
        float s = 0.0f;
#pragma unroll
        for (int t = 0; t < 16; ++t)
            if (t < t0)
                s += ct[t] * u[((((size_t)b * t0 + t) * t1 + uu) * z1 + z) * P + p];
        v_base[(size_t)b * v_sB + ((size_t)uu * P + p) * z1 + z] = (_Float16)(factor * s);
    }
}

// ---------------------------------------------------------------------------
// Routing step 2: logits[b,t,u,p] += sum_z u[b,t,u,z,p] * v[b,u,p,z]
// ---------------------------------------------------------------------------
__global__ void route_update(const float* __restrict__ u,
                             const _Float16* __restrict__ v_base, long v_sB,
                             float* __restrict__ logits,
                             int B, int t0, int t1, int z1, int P)
{
    size_t idx = (size_t)blockIdx.x * blockDim.x + threadIdx.x;
    size_t total = (size_t)B * t0 * t1 * P;
    if (idx >= total) return;
    int p   = (int)(idx % P);
    int rem = (int)(idx / P);
    int uu  = rem % t1; rem /= t1;
    int t   = rem % t0;
    int b   = rem / t0;

    float dot = 0.0f;
    for (int z = 0; z < z1; ++z) {
        float uv = u[((((size_t)b * t0 + t) * t1 + uu) * z1 + z) * P + p];
        float vv = (float)v_base[(size_t)b * v_sB + ((size_t)uu * P + p) * z1 + z];
        dot += uv * vv;
    }
    logits[(((size_t)b * t0 + t) * t1 + uu) * P + p] += dot;
}

// ---------------------------------------------------------------------------
// out[c,p] = sqrt(y[0,c,p]^2 + y[1,c,p]^2)   (L2 over batch dim, B=2)
// ---------------------------------------------------------------------------
__global__ void final_norm(const float* __restrict__ y, float* __restrict__ out,
                           size_t cp, long bStride)
{
    size_t i = (size_t)blockIdx.x * blockDim.x + threadIdx.x;
    if (i >= cp) return;
    float a = y[i];
    float b = y[(size_t)bStride + i];
    out[i] = sqrtf(a * a + b * b);
}

// ---------------------------------------------------------------------------
// Host-side drivers
// ---------------------------------------------------------------------------
static inline int cin_shift(int c) { return (c == 1) ? 0 : (c == 16) ? 4 : (c == 32) ? 5 : 6; }

static void launch_conv(hipStream_t s,
                        const _Float16* in, long in_sB, int t0, int Cin, int H, int W,
                        const float* w, const float* bias, _Float16* w16,
                        float* outF, _Float16* outH, long out_sB,
                        int Cout, int Ho, int Wo, int stride, int deconv, int B)
{
    int sh = cin_shift(Cin);
    int K = Cin * 25;
    int Kpad = (K + 31) & ~31;
    {
        int nw = Cout * Kpad;
        cvt_weights<<<(nw + 255) / 256, 256, 0, s>>>(w, w16, Cout, sh, Kpad);
    }
    dim3 g((Cout + 15) / 16, (Ho * Wo + 255) / 256, B * t0);
    if (deconv)
        conv5x5_wmma<1><<<g, 128, 0, s>>>(in, in_sB, t0, sh, H, W, w16, Kpad, bias,
                                          outF, outH, out_sB, Cout, Ho, Wo, stride);
    else
        conv5x5_wmma<0><<<g, 128, 0, s>>>(in, in_sB, t0, sh, H, W, w16, Kpad, bias,
                                          outF, outH, out_sB, Cout, Ho, Wo, stride);
}

static void run_caps_layer(hipStream_t s,
                           const _Float16* in_base, long in_sB, int t0, int z0, int H, int W,
                           const float* w, const float* bias,
                           int t1, int z1, int stride, int routing, int deconv,
                           _Float16* v_base, long v_sB,
                           float* U, float* L, _Float16* w16, int B)
{
    int Ho = deconv ? H * 2 : (stride == 2 ? H / 2 : H);
    int Wo = deconv ? W * 2 : (stride == 2 ? W / 2 : W);
    int P = Ho * Wo;
    int Cout = t1 * z1;

    launch_conv(s, in_base, in_sB, t0, z0, H, W, w, bias, w16,
                U, nullptr, (long)t0 * Cout * P, Cout, Ho, Wo, stride, deconv, B);

    size_t nl = (size_t)B * t0 * t1 * P;
    fill0<<<(unsigned)((nl + 255) / 256), 256, 0, s>>>(L, nl);

    for (int i = 0; i < routing; ++i) {
        size_t n1 = (size_t)B * t1 * P;
        route_reduce<<<(unsigned)((n1 + 255) / 256), 256, 0, s>>>(
            U, L, v_base, v_sB, B, t0, t1, z1, P);
        if (i + 1 < routing) {
            size_t n2 = (size_t)B * t0 * t1 * P;
            route_update<<<(unsigned)((n2 + 255) / 256), 256, 0, s>>>(
                U, v_base, v_sB, L, B, t0, t1, z1, P);
        }
    }
}

extern "C" void kernel_launch(void* const* d_in, const int* in_sizes, int n_in,
                              void* d_out, int out_size, void* d_ws, size_t ws_size,
                              hipStream_t stream)
{
    (void)in_sizes; (void)n_in; (void)out_size; (void)ws_size;

    const int B = 2;
    const int P384 = 384 * 384, P192 = 192 * 192, P96 = 96 * 96, P48 = 48 * 48;

    // ---- workspace bump allocator
    char* ws = (char*)d_ws;
    size_t off = 0;
    auto alloc_f = [&](size_t n) -> float* {
        float* p = (float*)(ws + off);
        off += ((n * sizeof(float)) + 255) & ~(size_t)255;
        return p;
    };
    auto alloc_h = [&](size_t n) -> _Float16* {
        _Float16* p = (_Float16*)(ws + off);
        off += ((n * sizeof(_Float16)) + 255) & ~(size_t)255;
        return p;
    };
    float*    U      = alloc_f(37748736);                    // max votes (d4/c5/c7/d8)
    float*    L      = alloc_f(2359296);                     // max logits
    float*    y2     = alloc_f((size_t)B * 10 * P384);       // conv2 output (fp32)
    _Float16* cat384 = alloc_h((size_t)B * 3  * 16 * P384);  // [b][t][p][z]
    _Float16* cat192 = alloc_h((size_t)B * 8  * 16 * P192);
    _Float16* cat96  = alloc_h((size_t)B * 16 * 32 * P96);
    _Float16* actA   = alloc_h((size_t)B * 16 * P384);       // ping activation
    _Float16* actB   = alloc_h((size_t)B * 8 * 32 * P48);    // pong (c3b out)
    _Float16* x16    = alloc_h((size_t)B * P384);
    _Float16* w16    = alloc_h((size_t)512 * 800);           // largest layer's weights

    const float* x = (const float*)d_in[0];
    auto W_ = [&](int i) { return (const float*)d_in[i]; };

    // stem input fp32 -> f16 (Cin=1, z-contig layout == planar)
    {
        size_t nx = (size_t)B * P384;
        cvt_f32_to_f16<<<(unsigned)((nx + 255) / 256), 256, 0, stream>>>(x, x16, nx);
    }

    // stem conv1 (no bias) -> f16 activation at cat384 @ t=2 (doubles as skip1)
    {
        cvt_weights<<<(16 * 32 + 255) / 256, 256, 0, stream>>>(W_(1), w16, 16, 0, 32);
        dim3 g(1, (P384 + 63) / 64, B);
        conv_stem_wmma<<<g, 128, 0, stream>>>(
            x16, w16, cat384 + (size_t)2 * 16 * P384, (long)3 * 16 * P384, 384, 384);
    }

    // c1a: (1,16) s2 -> (2,16) @192
    run_caps_layer(stream, cat384 + (size_t)2 * 16 * P384, (long)3 * 16 * P384,
                   1, 16, 384, 384, W_(2), W_(3), 2, 16, 2, 1, 0,
                   actA, (long)2 * 16 * P192, U, L, w16, B);
    // c1b: (2,16) -> (4,16) @192 -> cat192 @ t=4 (skip2)
    run_caps_layer(stream, actA, (long)2 * 16 * P192,
                   2, 16, 192, 192, W_(4), W_(5), 4, 16, 1, 3, 0,
                   cat192 + (size_t)4 * 16 * P192, (long)8 * 16 * P192, U, L, w16, B);
    // c2a: (4,16) s2 -> (4,32) @96
    run_caps_layer(stream, cat192 + (size_t)4 * 16 * P192, (long)8 * 16 * P192,
                   4, 16, 192, 192, W_(6), W_(7), 4, 32, 2, 3, 0,
                   actA, (long)4 * 32 * P96, U, L, w16, B);
    // c2b: (4,32) -> (8,32) @96 -> cat96 @ t=8 (skip3)
    run_caps_layer(stream, actA, (long)4 * 32 * P96,
                   4, 32, 96, 96, W_(8), W_(9), 8, 32, 1, 3, 0,
                   cat96 + (size_t)8 * 32 * P96, (long)16 * 32 * P96, U, L, w16, B);
    // c3a: (8,32) s2 -> (8,64) @48
    run_caps_layer(stream, cat96 + (size_t)8 * 32 * P96, (long)16 * 32 * P96,
                   8, 32, 96, 96, W_(10), W_(11), 8, 64, 2, 3, 0,
                   actA, (long)8 * 64 * P48, U, L, w16, B);
    // c3b: (8,64) -> (8,32) @48
    run_caps_layer(stream, actA, (long)8 * 64 * P48,
                   8, 64, 48, 48, W_(12), W_(13), 8, 32, 1, 3, 0,
                   actB, (long)8 * 32 * P48, U, L, w16, B);
    // d4: deconv (8,32) -> (8,32) @96 -> cat96 @ t=0
    run_caps_layer(stream, actB, (long)8 * 32 * P48,
                   8, 32, 48, 48, W_(14), W_(15), 8, 32, 2, 3, 1,
                   cat96, (long)16 * 32 * P96, U, L, w16, B);
    // c5: (16,32) -> (4,32) @96
    run_caps_layer(stream, cat96, (long)16 * 32 * P96,
                   16, 32, 96, 96, W_(16), W_(17), 4, 32, 1, 3, 0,
                   actA, (long)4 * 32 * P96, U, L, w16, B);
    // d6: deconv (4,32) -> (4,16) @192 -> cat192 @ t=0
    run_caps_layer(stream, actA, (long)4 * 32 * P96,
                   4, 32, 96, 96, W_(18), W_(19), 4, 16, 2, 3, 1,
                   cat192, (long)8 * 16 * P192, U, L, w16, B);
    // c7: (8,16) -> (4,16) @192
    run_caps_layer(stream, cat192, (long)8 * 16 * P192,
                   8, 16, 192, 192, W_(20), W_(21), 4, 16, 1, 3, 0,
                   actA, (long)4 * 16 * P192, U, L, w16, B);
    // d8: deconv (4,16) -> (2,16) @384 -> cat384 @ t=0
    run_caps_layer(stream, actA, (long)4 * 16 * P192,
                   4, 16, 192, 192, W_(22), W_(23), 2, 16, 2, 3, 1,
                   cat384, (long)3 * 16 * P384, U, L, w16, B);
    // c10: (3,16) -> (1,16) @384
    run_caps_layer(stream, cat384, (long)3 * 16 * P384,
                   3, 16, 384, 384, W_(24), W_(25), 1, 16, 1, 3, 0,
                   actA, (long)1 * 16 * P384, U, L, w16, B);

    // head: conv2 (16 -> 10, with bias) -> y2 fp32 [2,10,P]
    launch_conv(stream, actA, (long)16 * P384, 1, 16, 384, 384,
                W_(26), W_(27), w16, y2, nullptr, (long)10 * P384,
                10, 384, 384, 1, 0, B);

    // L2 over the batch dim -> [1,10,384,384]
    size_t cp = (size_t)10 * P384;
    final_norm<<<(unsigned)((cp + 255) / 256), 256, 0, stream>>>(
        y2, (float*)d_out, cp, (long)10 * P384);
}
